// Net_40570261078283
// MI455X (gfx1250) — matrix operand order repaired and checked by reading, not compiled, not run on hardware
//
#include <hip/hip_runtime.h>
#include <hip/hip_bf16.h>

#define NN   100000
#define EE   1600000
#define HIDC 128
#define GGC  100
#define EPSC 1e-5f

typedef __attribute__((ext_vector_type(16))) _Float16 v16h;
typedef __attribute__((ext_vector_type(8)))  _Float16 v8h;
typedef __attribute__((ext_vector_type(2)))  _Float16 v2h;
typedef __attribute__((ext_vector_type(8)))  float    v8f;

__device__ __forceinline__ v16h cat8(v8h lo, v8h hi) {
    return __builtin_shufflevector(lo, hi, 0,1,2,3,4,5,6,7,8,9,10,11,12,13,14,15);
}

__device__ __forceinline__ void atomic_add_f32(float* p, float v) {
    __hip_atomic_fetch_add(p, v, __ATOMIC_RELAXED, __HIP_MEMORY_SCOPE_AGENT);
}

// ---------------------------------------------------------------- zero scratch
__global__ void zero_k(float* __restrict__ p, long n) {
    long i = (long)blockIdx.x * blockDim.x + threadIdx.x;
    if (i < n) p[i] = 0.f;
}

// ------------------------------------------------- convert W2 weights to f16
__global__ void cvt_k(const float* __restrict__ w1, const float* __restrict__ w2,
                      _Float16* __restrict__ o1, _Float16* __restrict__ o2) {
    int i = blockIdx.x * blockDim.x + threadIdx.x;
    if (i < HIDC * HIDC) {
        o1[i] = (_Float16)w1[i];
        o2[i] = (_Float16)w2[i];
    }
}

// ------------------------------------------------------------- main edge pass
// Per block: 128 edges. 8 waves, each owns one 16-row M-tile.
// edge_attr[2] -> hidden[128] (f16 LDS) -> WMMA x [128x128] W2 -> atomic scatter.
__global__ void __launch_bounds__(256) edge_k(
    const float* __restrict__ edge_attr, const int* __restrict__ edge_index,
    const float* __restrict__ c1W1, const float* __restrict__ c1b1,
    const float* __restrict__ c2W1, const float* __restrict__ c2b1,
    const _Float16* __restrict__ W2a, const _Float16* __restrict__ W2b,
    float* __restrict__ acc1, float* __restrict__ acc2, float* __restrict__ cnt)
{
    __shared__ __align__(16) _Float16 hbuf[128 * 128];   // 32 KB hidden tile
    __shared__ float ea[256];
    __shared__ int   tgts[128];

    const int  t  = threadIdx.x;
    const long eb = (long)blockIdx.x * 128;
    const int* tgt = edge_index + EE;                    // edge_index[1]

    {   // stage edge_attr (128 edges x 2 f32) and targets
        long gi = eb * 2 + t;
        ea[t] = (gi < 2L * EE) ? edge_attr[gi] : 0.f;
    }
    bool evalid = false;
    if (t < 128) {
        evalid  = (eb + t) < EE;
        tgts[t] = evalid ? tgt[eb + t] : 0;   // clamp: invalid rows contribute 0.0
    }
    __syncthreads();

    const int  lane   = t & 31;
    const int  wave   = t >> 5;          // m-tile index 0..7
    const int  l15    = lane & 15;
    const bool lolane = lane < 16;

    for (int conv = 0; conv < 2; ++conv) {
        const float*     W1  = conv ? c2W1 : c1W1;
        const float*     B1  = conv ? c2b1 : c1b1;
        const _Float16*  W2  = conv ? W2b  : W2a;
        float*           acc = conv ? acc2 : acc1;

        // ---- first-layer MLP: hidden[e][k] = relu(a0*w0 + a1*w1 + b)
        // thread -> channel pair (k2, k2+1), edge stride 4; packed 32-bit LDS stores
        {
            int   k2 = (t & 63) * 2;
            int   e0 = t >> 6;
            float w0a = W1[2 * k2],     w1a = W1[2 * k2 + 1], ba = B1[k2];
            float w0b = W1[2 * k2 + 2], w1b = W1[2 * k2 + 3], bb = B1[k2 + 1];
            #pragma unroll 4
            for (int i = 0; i < 32; ++i) {
                int   e  = e0 + 4 * i;
                float a0 = ea[2 * e], a1 = ea[2 * e + 1];
                float va = fmaxf(fmaf(a0, w0a, fmaf(a1, w1a, ba)), 0.f);
                float vb = fmaxf(fmaf(a0, w0b, fmaf(a1, w1b, bb)), 0.f);
                v2h p; p[0] = (_Float16)va; p[1] = (_Float16)vb;
                *(v2h*)(hbuf + e * 128 + k2) = p;
            }
        }
        __syncthreads();

        // ---- A fragments: 16x32 f16 tiles, 4 k-steps (K=128)
        v16h A[4];
        {
            const _Float16* base = hbuf + (wave * 16 + l15) * 128 + (lolane ? 0 : 8);
            #pragma unroll
            for (int kk = 0; kk < 4; ++kk) {
                v8h lo = *(const v8h*)(base + kk * 32);
                v8h hi = *(const v8h*)(base + kk * 32 + 16);
                A[kk] = cat8(lo, hi);
            }
        }
        // ---- per-row accumulator offsets (D layout: M = r + (lane<16?0:8))
        int offr[8];
        #pragma unroll
        for (int r = 0; r < 8; ++r)
            offr[r] = tgts[wave * 16 + (lolane ? 0 : 8) + r] * HIDC + l15;

        // ---- N loop: 8 n-tiles of 16 output channels
        #pragma unroll 1
        for (int nt = 0; nt < 8; ++nt) {
            v8f cfrag = {};
            const _Float16* bbase = W2 + (nt * 16 + l15) * 128 + (lolane ? 0 : 8);
            #pragma unroll
            for (int kk = 0; kk < 4; ++kk) {
                v8h lo = *(const v8h*)(bbase + kk * 32);
                v8h hi = *(const v8h*)(bbase + kk * 32 + 16);
                v16h B = cat8(lo, hi);
                cfrag = __builtin_amdgcn_wmma_f32_16x16x32_f16(
                    false, A[kk], false, B, (short)0, cfrag, false, false);
            }
            // unconditional straight-line scatter (invalid rows add exact 0.0 to row 0)
            int ntoff = nt * 16;
            #pragma unroll
            for (int r = 0; r < 8; ++r)
                atomic_add_f32(acc + offr[r] + ntoff, cfrag[r]);
        }
        __syncthreads();   // before hbuf is overwritten for next conv
    }

    if (t < 128 && evalid)
        atomic_add_f32(cnt + tgts[t], 1.0f);
}

// ------------------------------- mean-aggregate + conv relu + BN stats (fused)
__global__ void bnstat_k(float* __restrict__ acc1, float* __restrict__ acc2,
                         const float* __restrict__ cnt,
                         const float* __restrict__ b21, const float* __restrict__ b22,
                         float* __restrict__ bsum, float* __restrict__ bsqs)
{
    int conv = blockIdx.y;
    float* acc      = conv ? acc2 : acc1;
    const float* b2 = conv ? b22  : b21;
    int c  = threadIdx.x;                 // channel
    int r0 = blockIdx.x * 128;
    int r1 = r0 + 128; if (r1 > NN) r1 = NN;
    float b = b2[c], s = 0.f, ss = 0.f;
    for (int r = r0; r < r1; ++r) {
        float cn = cnt[r];
        float v  = (cn > 0.f) ? fmaxf(acc[(size_t)r * HIDC + c] / cn + b, 0.f) : 0.f;
        acc[(size_t)r * HIDC + c] = v;    // in-place: acc now holds conv output
        s += v; ss += v * v;
    }
    atomic_add_f32(&bsum[conv * HIDC + c], s);
    atomic_add_f32(&bsqs[conv * HIDC + c], ss);
}

// -------------------------------------------- finalize BN affine + h0 row
__global__ void bnfin_k(const float* __restrict__ bsum, const float* __restrict__ bsqs,
                        const float* __restrict__ bn2g, const float* __restrict__ bn2b,
                        const float* __restrict__ bn3g, const float* __restrict__ bn3b,
                        const float* __restrict__ bn1b,
                        float* __restrict__ scale, float* __restrict__ shift,
                        float* __restrict__ h0row)
{
    int t = threadIdx.x;                  // 0..255  -> [conv][channel]
    int conv = t >> 7, c = t & 127;
    const float* g  = conv ? bn3g : bn2g;
    const float* be = conv ? bn3b : bn2b;
    float mu  = bsum[t] * (1.0f / NN);
    float var = bsqs[t] * (1.0f / NN) - mu * mu;
    float sc  = g[c] * rsqrtf(fmaxf(var, 0.f) + EPSC);
    scale[t] = sc;
    shift[t] = be[c] - mu * sc;
    if (t < HIDC) h0row[t] = fmaxf(bn1b[t], 0.f);
}

// ---------------------------------------- node_pool = h0 + relu(bn1) + relu(bn2)
__global__ void nodeout_k(const float* __restrict__ conv1, const float* __restrict__ conv2,
                          const float* __restrict__ scale, const float* __restrict__ shift,
                          const float* __restrict__ h0row, float* __restrict__ out)
{
    long i = (long)blockIdx.x * 256 + threadIdx.x;
    if (i < (long)NN * HIDC) {
        int c = (int)(i & 127);
        float v1 = fmaxf(scale[c]        * conv1[i] + shift[c],        0.f);
        float v2 = fmaxf(scale[HIDC + c] * conv2[i] + shift[HIDC + c], 0.f);
        out[i] = h0row[c] + v1 + v2;
    }
}

// ---------------------------------------- per-graph max/mean pooling (no atomics)
__device__ __forceinline__ int lowerb(const int* __restrict__ a, int n, int key) {
    int lo = 0, hi = n;
    while (lo < hi) { int mid = (lo + hi) >> 1; if (a[mid] < key) lo = mid + 1; else hi = mid; }
    return lo;
}

__global__ void pool_k(const float* __restrict__ conv1, const float* __restrict__ conv2,
                       const int* __restrict__ batch,
                       const float* __restrict__ scale, const float* __restrict__ shift,
                       float* __restrict__ gmax, float* __restrict__ gmean)
{
    int g = blockIdx.x;
    __shared__ int range[2];
    if (threadIdx.x == 0) {
        range[0] = lowerb(batch, NN, g);
        range[1] = lowerb(batch, NN, g + 1);
    }
    __syncthreads();
    int lo = range[0], hi = range[1];
    int c = threadIdx.x;                  // 128 threads: one channel each
    float s1 = scale[c], h1 = shift[c], s2 = scale[HIDC + c], h2 = shift[HIDC + c];
    float mx1 = -3.402823466e38f, mx2 = -3.402823466e38f, sm1 = 0.f, sm2 = 0.f;
    for (int r = lo; r < hi; ++r) {
        float v1 = fmaxf(s1 * conv1[(size_t)r * HIDC + c] + h1, 0.f);
        float v2 = fmaxf(s2 * conv2[(size_t)r * HIDC + c] + h2, 0.f);
        mx1 = fmaxf(mx1, v1); sm1 += v1;
        mx2 = fmaxf(mx2, v2); sm2 += v2;
    }
    float inv = 1.0f / fmaxf((float)(hi - lo), 1.0f);
    gmax [g * HIDC + c]               = mx1;
    gmean[g * HIDC + c]               = sm1 * inv;
    gmax [GGC * HIDC + g * HIDC + c]  = mx2;
    gmean[GGC * HIDC + g * HIDC + c]  = sm2 * inv;
}

// ------------------------------------------------- head: gp = sum_l pool_l @ Wp_l.T
__global__ void head_k(const float* __restrict__ gmax, const float* __restrict__ gmean,
                       const float* __restrict__ h0row,
                       const float* __restrict__ Wp0, const float* __restrict__ bp0,
                       const float* __restrict__ Wp1, const float* __restrict__ bp1,
                       const float* __restrict__ Wp2, const float* __restrict__ bp2,
                       float* __restrict__ out)
{
    int i = blockIdx.x * 256 + threadIdx.x;
    if (i >= GGC * HIDC) return;
    int g = i >> 7, c = i & 127;
    const float* w0 = Wp0 + c * 2 * HIDC;
    const float* w1 = Wp1 + c * 2 * HIDC;
    const float* w2 = Wp2 + c * 2 * HIDC;
    const float* mx1 = gmax  + g * HIDC;
    const float* mn1 = gmean + g * HIDC;
    const float* mx2 = gmax  + GGC * HIDC + g * HIDC;
    const float* mn2 = gmean + GGC * HIDC + g * HIDC;
    float s = bp0[c] + bp1[c] + bp2[c];
    #pragma unroll 4
    for (int k = 0; k < HIDC; ++k) {
        float h0 = h0row[k];
        s = fmaf(h0,     w0[k] + w0[HIDC + k], s);     // pool(h0) = [h0row, h0row]
        s = fmaf(mx1[k], w1[k],        s);
        s = fmaf(mn1[k], w1[HIDC + k], s);
        s = fmaf(mx2[k], w2[k],        s);
        s = fmaf(mn2[k], w2[HIDC + k], s);
    }
    out[i] = s;
}

// ------------------------------------------------------------------- launcher
extern "C" void kernel_launch(void* const* d_in, const int* in_sizes, int n_in,
                              void* d_out, int out_size, void* d_ws, size_t ws_size,
                              hipStream_t stream) {
    const float* edge_attr = (const float*)d_in[1];
    const float* bn1_b     = (const float*)d_in[5];
    const float* bn2_g     = (const float*)d_in[6];
    const float* bn2_b     = (const float*)d_in[7];
    const float* bn3_g     = (const float*)d_in[8];
    const float* bn3_b     = (const float*)d_in[9];
    const float* c1_W1     = (const float*)d_in[10];
    const float* c1_b1     = (const float*)d_in[11];
    const float* c1_W2     = (const float*)d_in[12];
    const float* c1_b2     = (const float*)d_in[13];
    const float* c2_W1     = (const float*)d_in[14];
    const float* c2_b1     = (const float*)d_in[15];
    const float* c2_W2     = (const float*)d_in[16];
    const float* c2_b2     = (const float*)d_in[17];
    const float* Wp0       = (const float*)d_in[18];
    const float* bp0       = (const float*)d_in[19];
    const float* Wp1       = (const float*)d_in[20];
    const float* bp1       = (const float*)d_in[21];
    const float* Wp2       = (const float*)d_in[22];
    const float* bp2       = (const float*)d_in[23];
    const int*   edge_index = (const int*)d_in[24];
    const int*   batch      = (const int*)d_in[25];

    // workspace layout (all 128B-aligned by construction)
    float* acc1  = (float*)d_ws;                       // [NN, HIDC]
    float* acc2  = acc1  + (size_t)NN * HIDC;          // [NN, HIDC]
    float* cnt   = acc2  + (size_t)NN * HIDC;          // [NN]
    float* bsum  = cnt   + NN;                         // [2,128]
    float* bsqs  = bsum  + 256;                        // [2,128]
    float* scale = bsqs  + 256;                        // [2,128]
    float* shift = scale + 256;                        // [2,128]
    float* h0row = shift + 256;                        // [128] (+128 pad)
    float* gmax  = h0row + 256;                        // [2,GGC,HIDC]
    float* gmean = gmax  + 2 * GGC * HIDC;             // [2,GGC,HIDC]
    _Float16* W2h = (_Float16*)(gmean + 2 * GGC * HIDC); // 2x[128,128] f16

    float* out_nodes = (float*)d_out;
    float* out_graph = out_nodes + (size_t)NN * HIDC;

    // 1) zero accumulators + stats
    long nz = 2L * NN * HIDC + NN + 512;
    zero_k<<<(int)((nz + 255) / 256), 256, 0, stream>>>(acc1, nz);

    // 2) convert W2 weights to f16
    cvt_k<<<(HIDC * HIDC + 255) / 256, 256, 0, stream>>>(c1_W2, c2_W2, W2h, W2h + HIDC * HIDC);

    // 3) fused edge MLP + WMMA GEMM + scatter (both convs)
    edge_k<<<EE / 128, 256, 0, stream>>>(edge_attr, edge_index,
                                         c1_W1, c1_b1, c2_W1, c2_b1,
                                         W2h, W2h + HIDC * HIDC,
                                         acc1, acc2, cnt);

    // 4) mean + relu + BN statistics (in-place)
    bnstat_k<<<dim3((NN + 127) / 128, 2), 128, 0, stream>>>(acc1, acc2, cnt,
                                                            c1_b2, c2_b2, bsum, bsqs);

    // 5) BN affine params + h0 row
    bnfin_k<<<1, 256, 0, stream>>>(bsum, bsqs, bn2_g, bn2_b, bn3_g, bn3_b, bn1_b,
                                   scale, shift, h0row);

    // 6) node_pool output
    nodeout_k<<<(int)(((long)NN * HIDC + 255) / 256), 256, 0, stream>>>(
        acc1, acc2, scale, shift, h0row, out_nodes);

    // 7) per-graph pooling
    pool_k<<<GGC, 128, 0, stream>>>(acc1, acc2, batch, scale, shift, gmax, gmean);

    // 8) head
    head_k<<<(GGC * HIDC + 255) / 256, 256, 0, stream>>>(gmax, gmean, h0row,
                                                         Wp0, bp0, Wp1, bp1, Wp2, bp2,
                                                         out_graph);
}
